// RobustSS2DFusion_54022098649129
// MI455X (gfx1250) — compile-verified, matrix operand order checked by reading, not run on hardware
//
#include <hip/hip_runtime.h>
#include <hip/hip_bf16.h>
#include <math.h>

// ---- problem constants (match reference) ----
#define B_SCENES 2
#define KN 5
#define CH 128
#define HH 64
#define WW 64
#define HW 4096            // H*W
#define LL (KN * HW)       // 20480  (SS2D sequence length: Hs=K, Ws=HW)
#define DSTATE 16
#define DTRANK 8
#define XDROWS 48          // 40 rows of x_dbl padded to 3 WMMA tiles
#define SCAN_TILE 64       // columns of x_dbl staged into LDS per chunk

typedef __attribute__((ext_vector_type(16))) _Float16 v16h;
typedef __attribute__((ext_vector_type(8)))  float    v8f;

// -----------------------------------------------------------------------------
// WMMA 16x16 tile GEMM helper: D = A(16xKdim) * B(Kdimx16), f16 in / f32 acc.
// Fragment layouts per CDNA5 ISA 7.12.2.
// -----------------------------------------------------------------------------
template <typename FA, typename FB>
__device__ __forceinline__ v8f wmma_tile_f16(int kdim, FA fa, FB fb) {
    v8f acc = {};
    const int lane = threadIdx.x & 31;
    const int half = lane >> 4;
    const int idx  = lane & 15;
    for (int k0 = 0; k0 < kdim; k0 += 32) {
        v16h A, Bv;
#pragma unroll
        for (int p = 0; p < 8; ++p) {
            int kk = k0 + (p < 4 ? 2 * p : 16 + 2 * (p - 4)) + 8 * half;
            A[2 * p]      = (_Float16)fa(idx, kk);
            A[2 * p + 1]  = (_Float16)fa(idx, kk + 1);
            Bv[2 * p]     = (_Float16)fb(kk, idx);
            Bv[2 * p + 1] = (_Float16)fb(kk + 1, idx);
        }
        acc = __builtin_amdgcn_wmma_f32_16x16x32_f16(
            /*neg_a=*/false, A, /*neg_b=*/false, Bv,
            /*c_mod=*/(short)0, acc, /*reuse_a=*/false, /*reuse_b=*/false);
    }
    return acc;
}

// -----------------------------------------------------------------------------
// K1: bilinear warp. grid = B*K*HW blocks, block = 128 threads (one channel).
// -----------------------------------------------------------------------------
__global__ void k_warp(const float* __restrict__ x, const float* __restrict__ nam,
                       float* __restrict__ warped) {
    int pix = blockIdx.x;              // b*K*HW + k*HW + p
    int c   = threadIdx.x;
    int p   = pix & (HW - 1);
    int bk  = pix >> 12;               // b*K + k
    int b   = bk / KN, k = bk % KN;
    const float* th = nam + ((size_t)(b * KN + 0) * KN + k) * 6;  // nam[b,0,k]
    int px = p & (WW - 1), py = p >> 6;
    float gx = (px + 0.5f) * (2.0f / WW) - 1.0f;
    float gy = (py + 0.5f) * (2.0f / HH) - 1.0f;
    float u = th[0] * gx + th[1] * gy + th[2];
    float v = th[3] * gx + th[4] * gy + th[5];
    float ix = ((u + 1.0f) * WW - 1.0f) * 0.5f;
    float iy = ((v + 1.0f) * HH - 1.0f) * 0.5f;
    float x0 = floorf(ix), y0 = floorf(iy);
    float wx1 = ix - x0, wy1 = iy - y0;
    const float* src = x + ((size_t)bk * CH + c) * HW;
    float acc = 0.f;
#pragma unroll
    for (int dy = 0; dy < 2; ++dy)
#pragma unroll
        for (int dx = 0; dx < 2; ++dx) {
            float xf = x0 + dx, yf = y0 + dy;
            float wt = (dx ? wx1 : 1.f - wx1) * (dy ? wy1 : 1.f - wy1);
            bool ok = (xf >= 0.f) && (xf <= (float)(WW - 1)) &&
                      (yf >= 0.f) && (yf <= (float)(HH - 1));
            int xc = (int)fminf(fmaxf(xf, 0.f), (float)(WW - 1));
            int yc = (int)fminf(fmaxf(yf, 0.f), (float)(HH - 1));
            float val = src[yc * WW + xc];
            acc += ok ? wt * val : 0.f;
        }
    warped[((size_t)bk * CH + c) * HW + p] = acc;
}

// -----------------------------------------------------------------------------
// K2: robust weights + gate + softmax over K.  One wave32 per pixel.
// -----------------------------------------------------------------------------
__global__ void k_alpha(const float* __restrict__ warped, const float* __restrict__ nam,
                        float* __restrict__ alpha) {
    int wid  = (int)((blockIdx.x * (size_t)blockDim.x + threadIdx.x) >> 5);
    int lane = threadIdx.x & 31;
    if (wid >= B_SCENES * HW) return;
    int b = wid >> 12, p = wid & (HW - 1);
    const float* wb = warped + (size_t)b * KN * CH * HW;

    float e[4];
#pragma unroll
    for (int j = 0; j < 4; ++j)
        e[j] = wb[((size_t)0 * CH + (lane * 4 + j)) * HW + p];

    float dotv[KN], s2[KN];
#pragma unroll
    for (int k = 0; k < KN; ++k) {
        float d = 0.f, s = 0.f;
#pragma unroll
        for (int j = 0; j < 4; ++j) {
            float v = wb[((size_t)k * CH + (lane * 4 + j)) * HW + p];
            d += e[j] * v;
            s += v * v;
        }
#pragma unroll
        for (int m = 16; m > 0; m >>= 1) {
            d += __shfl_xor(d, m, 32);
            s += __shfl_xor(s, m, 32);
        }
        dotv[k] = d;
        s2[k]   = s;
    }
    float ne = fmaxf(sqrtf(s2[0]), 1e-6f);
    float lg[KN];
#pragma unroll
    for (int k = 0; k < KN; ++k) {
        float nk  = fmaxf(sqrtf(s2[k]), 1e-6f);
        float sim = fminf(fmaxf(dotv[k] / (ne * nk), -1.f), 1.f);
        float r   = fmaxf(1.f - sim, 0.f);
        float xr  = r * (1.0f / 0.3f);                 // C_TUKEY
        float t   = fmaxf(1.f - xr * xr, 0.f);
        float wp  = fmaxf(t * t, 1e-6f);
        const float* th = nam + ((size_t)(b * KN) * KN + k) * 6;
        float tx  = th[2], ty = th[5];
        float yaw = fabsf(atan2f(th[1], th[0]));
        float z   = 2.0f - 0.4f * sqrtf(tx * tx + ty * ty) - 0.8f * yaw;
        float g   = 1.f / (1.f + expf(-z));
        lg[k]     = logf(wp + 1e-8f) + logf(g + 1e-8f);
    }
    float mx = lg[0];
#pragma unroll
    for (int k = 1; k < KN; ++k) mx = fmaxf(mx, lg[k]);
    float se = 0.f;
#pragma unroll
    for (int k = 0; k < KN; ++k) { lg[k] = expf(lg[k] - mx); se += lg[k]; }
    if (lane < KN)
        alpha[((size_t)b * KN + lane) * HW + p] = lg[lane] / se;
}

// -----------------------------------------------------------------------------
// K3: xin = in_proj_w (128x128) @ tokens (128xL), tokens fused on-the-fly.
// -----------------------------------------------------------------------------
__global__ void k_inproj(const float* __restrict__ warped, const float* __restrict__ alpha,
                         const float* __restrict__ wproj, float* __restrict__ xin) {
    const int NT = LL / 16;  // 1280
    int wave = (int)((blockIdx.x * (size_t)blockDim.x + threadIdx.x) >> 5);
    if (wave >= B_SCENES * 8 * NT) return;
    int b   = wave / (8 * NT);
    int rem = wave % (8 * NT);
    int m0  = (rem / NT) * 16;
    int n0  = (rem % NT) * 16;
    const float* wb = warped + (size_t)b * KN * CH * HW;
    const float* ab = alpha + (size_t)b * KN * HW;
    auto fa = [&](int m, int k) { return wproj[(size_t)(m0 + m) * CH + k]; };
    auto fb = [&](int k, int n) {
        int l = n0 + n;
        int kk = l >> 12, p = l & (HW - 1);
        return wb[((size_t)kk * CH + k) * HW + p] * ab[(size_t)kk * HW + p];
    };
    v8f acc = wmma_tile_f16(CH, fa, fb);
    float* D = xin + (size_t)b * CH * LL;
    int lane = threadIdx.x & 31, half = lane >> 4, col = lane & 15;
#pragma unroll
    for (int v = 0; v < 8; ++v)
        D[(size_t)(m0 + v + 8 * half) * LL + n0 + col] = acc[v];
}

// -----------------------------------------------------------------------------
// K4: depthwise 3x3 conv over (5 x 4096) + bias + SiLU, writes hw and wh layouts.
// -----------------------------------------------------------------------------
__global__ void k_conv(const float* __restrict__ xin, const float* __restrict__ cw,
                       const float* __restrict__ cb, float* __restrict__ xa,
                       float* __restrict__ xwh) {
    size_t i = (size_t)blockIdx.x * blockDim.x + threadIdx.x;
    if (i >= (size_t)B_SCENES * CH * LL) return;
    int l  = (int)(i % LL);
    int dc = (int)((i / LL) % CH);
    int b  = (int)(i / ((size_t)LL * CH));
    int h  = l >> 12;          // row in 0..4 (Ws = 4096)
    int w2 = l & (HW - 1);
    const float* src = xin + ((size_t)b * CH + dc) * LL;
    const float* wk  = cw + (size_t)dc * 9;
    float acc = cb[dc];
#pragma unroll
    for (int dy = -1; dy <= 1; ++dy)
#pragma unroll
        for (int dx = -1; dx <= 1; ++dx) {
            int hh = h + dy, ww2 = w2 + dx;
            float v = (hh >= 0 && hh < KN && ww2 >= 0 && ww2 < HW)
                          ? src[hh * HW + ww2] : 0.f;
            acc += wk[(dy + 1) * 3 + (dx + 1)] * v;
        }
    float s = acc / (1.f + expf(-acc));  // SiLU
    xa[((size_t)b * CH + dc) * LL + l] = s;
    xwh[((size_t)b * CH + dc) * LL + (size_t)w2 * KN + h] = s;
}

// -----------------------------------------------------------------------------
// K5: x_dbl[k] = x_proj_w[k] (40x128, padded 48) @ xs_k (128xL).
// Output stored L-MAJOR: xdbl[bk][l][48] so the scan reads contiguous columns.
// -----------------------------------------------------------------------------
__global__ void k_xproj(const float* __restrict__ xa, const float* __restrict__ xwh,
                        const float* __restrict__ xpw, float* __restrict__ xdbl) {
    const int NT  = LL / 16;
    const int PER = 3 * NT;
    int wave = (int)((blockIdx.x * (size_t)blockDim.x + threadIdx.x) >> 5);
    int bk   = wave / PER;
    if (bk >= B_SCENES * 4) return;
    int rem = wave % PER;
    int b = bk >> 2, kd = bk & 3;
    int m0 = (rem / NT) * 16;
    int n0 = (rem % NT) * 16;
    const float* xb = ((kd & 1) ? xwh : xa) + (size_t)b * CH * LL;
    auto fa = [&](int m, int k) {
        int r = m0 + m;
        return (r < 40) ? xpw[((size_t)kd * 40 + r) * CH + k] : 0.f;
    };
    auto fb = [&](int k, int n) {
        int l = n0 + n;
        int s = (kd < 2) ? l : (LL - 1 - l);
        return xb[(size_t)k * LL + s];
    };
    v8f acc = wmma_tile_f16(CH, fa, fb);
    float* D = xdbl + (size_t)(b * 4 + kd) * LL * XDROWS;
    int lane = threadIdx.x & 31, half = lane >> 4, col = lane & 15;
#pragma unroll
    for (int v = 0; v < 8; ++v)
        D[(size_t)(n0 + col) * XDROWS + m0 + v + 8 * half] = acc[v];
}

__global__ void k_zero(float* __restrict__ p, size_t n) {
    size_t i = (size_t)blockIdx.x * blockDim.x + threadIdx.x;
    if (i < n) p[i] = 0.f;
}

// -----------------------------------------------------------------------------
// K6: selective scan. One lane per (scene, direction, channel).
// x_dbl columns (dt-rank 8 | B 16 | C 16, padded 48) staged to LDS in 64-column
// chunks via the Tensor Data Mover (TDM) when available, else a cooperative
// float4 copy. All 128 lanes broadcast-read each staged column.
// -----------------------------------------------------------------------------
#define HAVE_TDM (__has_builtin(__builtin_amdgcn_tensor_load_to_lds) && \
                  __has_builtin(__builtin_amdgcn_s_wait_tensorcnt))
#if HAVE_TDM
typedef unsigned int u32x4 __attribute__((ext_vector_type(4)));
typedef int          i32x8 __attribute__((ext_vector_type(8)));
typedef int          i32x4 __attribute__((ext_vector_type(4)));
#endif

__global__ void k_scan(const float* __restrict__ xa, const float* __restrict__ xwh,
                       const float* __restrict__ xdbl, const float* __restrict__ dtw,
                       const float* __restrict__ dtb, const float* __restrict__ alog,
                       const float* __restrict__ dsv, float* __restrict__ y) {
    __shared__ __align__(16) float sdbl[SCAN_TILE * XDROWS];  // 12 KB

    int b  = blockIdx.x >> 2;
    int kd = blockIdx.x & 3;
    int d  = threadIdx.x;

    float Aa[DSTATE], h[DSTATE], wdt[DTRANK];
#pragma unroll
    for (int n = 0; n < DSTATE; ++n) {
        Aa[n] = -expf(alog[((size_t)kd * CH + d) * DSTATE + n]);
        h[n]  = 0.f;
    }
#pragma unroll
    for (int r = 0; r < DTRANK; ++r)
        wdt[r] = dtw[((size_t)kd * CH + d) * DTRANK + r];
    float bdt = dtb[(size_t)kd * CH + d];
    float Dv  = dsv[(size_t)kd * CH + d];

    const float* xb = ((kd & 1) ? xwh : xa) + ((size_t)b * CH + d) * LL;
    const float* xd = xdbl + (size_t)(b * 4 + kd) * LL * XDROWS;
    float* yb = y + (size_t)b * CH * HW + (size_t)d * HW;

    const int NELEM = SCAN_TILE * XDROWS;  // 3072 floats per chunk

    for (int l0 = 0; l0 < LL; l0 += SCAN_TILE) {
        __syncthreads();  // previous chunk fully consumed
#if HAVE_TDM
        if (threadIdx.x < 32) {  // wave 0 issues the tensor DMA (TDM ignores EXEC)
            unsigned lds_off = (unsigned)(size_t)(void*)sdbl;       // LDS byte offset
            unsigned long long ga =
                (unsigned long long)(size_t)(xd + (size_t)l0 * XDROWS);
            u32x4 g0;
            g0[0] = 1u;                                      // count=1, user mode
            g0[1] = lds_off;                                 // lds_addr
            g0[2] = (unsigned)(ga & 0xFFFFFFFFu);            // global_addr[31:0]
            g0[3] = (unsigned)((ga >> 32) & 0x01FFFFFFu)     // global_addr[56:32]
                    | 0x80000000u;                           // type=2 ("image")
            i32x8 g1;
            g1[0] = 0x20000;                 // workgroup_mask=0, data_size=4B
            g1[1] = (NELEM & 0xFFFF) << 16;  // tensor_dim0[15:0]
            g1[2] = ((NELEM >> 16) & 0xFFFF) | (1 << 16);  // dim0[31:16], dim1=1
            g1[3] = (NELEM << 16);           // dim1 hi=0, tile_dim0=3072
            g1[4] = 0;                       // tile_dim1=0 (1-D), tile_dim2=0
            g1[5] = NELEM;                   // tensor_dim0_stride lo32
            g1[6] = 0;
            g1[7] = 0;
            i32x4 gz4 = {0, 0, 0, 0};
            i32x8 gz8 = {0, 0, 0, 0, 0, 0, 0, 0};
            // amdgpu-toolchain (clang-23) 6-arg form:
            // (u32x4 g0, i32x8 g1, i32x4 g2, i32x4 g3, i32x8 extra, i32 cpol)
            __builtin_amdgcn_tensor_load_to_lds(g0, g1, gz4, gz4, gz8, 0);
            __builtin_amdgcn_s_wait_tensorcnt(0);
        }
#else
        const float4* s4 = (const float4*)(xd + (size_t)l0 * XDROWS);
        float4* d4 = (float4*)sdbl;
        for (int i = threadIdx.x; i < NELEM / 4; i += blockDim.x) d4[i] = s4[i];
#endif
        __syncthreads();

        // prefetch the per-channel x stream for the next chunk
        {
            int sn = (kd < 2) ? (l0 + SCAN_TILE) : (LL - 1 - (l0 + SCAN_TILE));
            if (sn >= 0 && sn < LL) __builtin_prefetch(xb + sn, 0, 1);
        }

        for (int t = 0; t < SCAN_TILE; ++t) {
            int l = l0 + t;
            const float* col = sdbl + t * XDROWS;
            float dtr = bdt;
#pragma unroll
            for (int r = 0; r < DTRANK; ++r) dtr += wdt[r] * col[r];
            float dt = (dtr > 20.f) ? dtr : log1pf(expf(dtr));  // softplus
            int s = (kd < 2) ? l : (LL - 1 - l);
            float xv = xb[s];
            float dx = dt * xv;
            float acc = 0.f;
#pragma unroll
            for (int n = 0; n < DSTATE; ++n) {
                float Bv = col[8 + n];
                float Cv = col[24 + n];
                h[n] = h[n] * __expf(dt * Aa[n]) + dx * Bv;
                acc += h[n] * Cv;
            }
            acc += Dv * xv;
            int dest;
            if (kd == 0)      dest = l;
            else if (kd == 2) dest = LL - 1 - l;
            else {
                int m = (kd == 1) ? l : (LL - 1 - l);   // wh-layout position
                dest = (m % KN) * HW + m / KN;          // untranspose to hw layout
            }
            if (dest < HW) atomicAdd(&yb[dest], acc);
        }
    }
}

// -----------------------------------------------------------------------------
// K7: channel LayerNorm over the 4096 k=0 pixels. One wave per pixel.
// -----------------------------------------------------------------------------
__global__ void k_ln(const float* __restrict__ y, const float* __restrict__ g,
                     const float* __restrict__ bb, float* __restrict__ yn) {
    int wid  = (int)((blockIdx.x * (size_t)blockDim.x + threadIdx.x) >> 5);
    int lane = threadIdx.x & 31;
    if (wid >= B_SCENES * HW) return;
    int b = wid >> 12, p = wid & (HW - 1);
    const float* yb = y + (size_t)b * CH * HW + p;
    float v[4];
    float s = 0.f;
#pragma unroll
    for (int j = 0; j < 4; ++j) {
        v[j] = yb[(size_t)(lane * 4 + j) * HW];
        s += v[j];
    }
#pragma unroll
    for (int m = 16; m > 0; m >>= 1) s += __shfl_xor(s, m, 32);
    float mu = s * (1.0f / CH);
    float q = 0.f;
#pragma unroll
    for (int j = 0; j < 4; ++j) { float dd = v[j] - mu; q += dd * dd; }
#pragma unroll
    for (int m = 16; m > 0; m >>= 1) q += __shfl_xor(q, m, 32);
    float inv = rsqrtf(q * (1.0f / CH) + 1e-5f);
    float* o = yn + ((size_t)b * HW + p) * CH;
#pragma unroll
    for (int j = 0; j < 4; ++j) {
        int c = lane * 4 + j;
        o[c] = (v[j] - mu) * inv * g[c] + bb[c];
    }
}

// -----------------------------------------------------------------------------
// K8: out[p, co] = yn(p,:) . out_proj_w(co,:), writes d_out[b][co][p] directly.
// -----------------------------------------------------------------------------
__global__ void k_outproj(const float* __restrict__ yn, const float* __restrict__ wo,
                          float* __restrict__ out) {
    const int MT  = HW / 16;  // 256 pixel tiles
    const int NTt = CH / 16;  // 8 channel tiles
    int wave = (int)((blockIdx.x * (size_t)blockDim.x + threadIdx.x) >> 5);
    int b = wave / (MT * NTt);
    if (b >= B_SCENES) return;
    int rem = wave % (MT * NTt);
    int m0 = (rem / NTt) * 16;
    int n0 = (rem % NTt) * 16;
    const float* yb = yn + (size_t)b * HW * CH;
    auto fa = [&](int m, int k) { return yb[(size_t)(m0 + m) * CH + k]; };
    auto fb = [&](int k, int n) { return wo[(size_t)(n0 + n) * CH + k]; };
    v8f acc = wmma_tile_f16(CH, fa, fb);
    float* ob = out + (size_t)b * CH * HW;
    int lane = threadIdx.x & 31, half = lane >> 4, col = lane & 15;
#pragma unroll
    for (int v = 0; v < 8; ++v)
        ob[(size_t)(n0 + col) * HW + m0 + v + 8 * half] = acc[v];
}

// -----------------------------------------------------------------------------
extern "C" void kernel_launch(void* const* d_in, const int* in_sizes, int n_in,
                              void* d_out, int out_size, void* d_ws, size_t ws_size,
                              hipStream_t stream) {
    const float* x    = (const float*)d_in[0];
    // d_in[1] = record_len (unused; always K)
    const float* nam  = (const float*)d_in[2];
    const float* inw  = (const float*)d_in[3];
    const float* cw   = (const float*)d_in[4];
    const float* cb   = (const float*)d_in[5];
    const float* xpw  = (const float*)d_in[6];
    const float* dtw  = (const float*)d_in[7];
    const float* dtb  = (const float*)d_in[8];
    const float* alog = (const float*)d_in[9];
    const float* dsv  = (const float*)d_in[10];
    const float* lng  = (const float*)d_in[11];
    const float* lnb  = (const float*)d_in[12];
    const float* wo   = (const float*)d_in[13];
    float* out = (float*)d_out;
    float* ws  = (float*)d_ws;

    // workspace layout (floats)
    const size_t SZ_WARPED = (size_t)B_SCENES * KN * CH * HW;   // 5,242,880
    const size_t SZ_ALPHA  = (size_t)B_SCENES * KN * HW;        //    40,960
    const size_t SZ_ACT    = (size_t)B_SCENES * CH * LL;        // 5,242,880
    const size_t SZ_XDBL   = (size_t)B_SCENES * 4 * XDROWS * LL;// 7,864,320
    const size_t SZ_Y      = (size_t)B_SCENES * CH * HW;        // 1,048,576
    float* warped = ws;
    float* alpha  = warped + SZ_WARPED;
    float* xin    = alpha + SZ_ALPHA;
    float* xa     = xin + SZ_ACT;
    float* xwh    = xa + SZ_ACT;
    float* xdbl   = xwh + SZ_ACT;
    float* ycomb  = xdbl + SZ_XDBL;
    float* yn     = ycomb + SZ_Y;

    const int NT = LL / 16;  // 1280

    k_warp<<<B_SCENES * KN * HW, CH, 0, stream>>>(x, nam, warped);
    k_alpha<<<(B_SCENES * HW) / 8, 256, 0, stream>>>(warped, nam, alpha);
    k_inproj<<<(B_SCENES * 8 * NT) / 4, 128, 0, stream>>>(warped, alpha, inw, xin);
    k_conv<<<(int)((SZ_ACT + 255) / 256), 256, 0, stream>>>(xin, cw, cb, xa, xwh);
    k_xproj<<<(B_SCENES * 4 * 3 * NT) / 4, 128, 0, stream>>>(xa, xwh, xpw, xdbl);
    k_zero<<<(int)((SZ_Y + 255) / 256), 256, 0, stream>>>(ycomb, SZ_Y);
    k_scan<<<B_SCENES * 4, CH, 0, stream>>>(xa, xwh, xdbl, dtw, dtb, alog, dsv, ycomb);
    k_ln<<<(B_SCENES * HW) / 8, 256, 0, stream>>>(ycomb, lng, lnb, yn);
    k_outproj<<<(B_SCENES * (HW / 16) * (CH / 16)) / 4, 128, 0, stream>>>(yn, wo, out);
}
